// AMPGCN_19490561589348
// MI455X (gfx1250) — matrix operand order repaired and checked by reading, not compiled, not run on hardware
//
#include <hip/hip_runtime.h>
#include <hip/hip_bf16.h>

typedef __attribute__((ext_vector_type(2)))  float    v2f;
typedef __attribute__((ext_vector_type(8)))  float    v8f;
typedef __attribute__((ext_vector_type(16))) _Float16 v16h;
typedef __attribute__((__vector_size__(16))) int      v4i;   // b128 payload type for async builtin

#define FTOK 32      // tokens per node
#define DDIM 6       // token dim
#define HID  192     // 32*6
#define NCLS 16

#define GLOBAL_AS __attribute__((address_space(1)))
#define LDS_AS    __attribute__((address_space(3)))

#if __has_builtin(__builtin_amdgcn_global_load_async_to_lds_b128)
#define AMPGCN_HAVE_ASYNC 1
#endif

// ---------------------------------------------------------------- zero
__global__ void ampgcn_zero(float* __restrict__ p, int n) {
    int i = blockIdx.x * blockDim.x + threadIdx.x;
    if (i < n) p[i] = 0.f;
}

// ---------------------------------------------------------------- per-node Q/K/V projection
// tok[n,f] = [feat_emb[f,0..4], x[n,f]*val_w[f]+val_b[f]]   (first 5 dims node-independent)
// q/k/v[n,f,:] = tok @ W + b, stored f16, row padded to 8 halves (16B) for b128 loads.
__global__ void ampgcn_node_proj(const float* __restrict__ x,
                                 const float* __restrict__ fe,
                                 const float* __restrict__ vw,
                                 const float* __restrict__ vb,
                                 const float* __restrict__ Wq, const float* __restrict__ Wk,
                                 const float* __restrict__ Wv,
                                 const float* __restrict__ bq, const float* __restrict__ bk,
                                 const float* __restrict__ bv,
                                 _Float16* __restrict__ qn, _Float16* __restrict__ kn,
                                 _Float16* __restrict__ vn, int Nn) {
    int idx = blockIdx.x * blockDim.x + threadIdx.x;
    if (idx >= Nn * FTOK) return;
    int f = idx & (FTOK - 1);
    float tok[DDIM];
#pragma unroll
    for (int j = 0; j < 5; ++j) tok[j] = fe[f * 5 + j];
    tok[5] = x[idx] * vw[f] + vb[f];

    _Float16* q = qn + (size_t)idx * 8;
    _Float16* k = kn + (size_t)idx * 8;
    _Float16* v = vn + (size_t)idx * 8;
#pragma unroll
    for (int d = 0; d < DDIM; ++d) {
        float aq = bq[d], ak = bk[d], av = bv[d];
#pragma unroll
        for (int j = 0; j < DDIM; ++j) {
            aq += tok[j] * Wq[j * DDIM + d];
            ak += tok[j] * Wk[j * DDIM + d];
            av += tok[j] * Wv[j * DDIM + d];
        }
        q[d] = (_Float16)aq; k[d] = (_Float16)ak; v[d] = (_Float16)av;
    }
    q[6] = q[7] = (_Float16)0.f;
    k[6] = k[7] = (_Float16)0.f;
    v[6] = v[7] = (_Float16)0.f;
}

// ---------------------------------------------------------------- per-edge attention (1 wave = 1 edge)
__global__ __launch_bounds__(32)
void ampgcn_edge_attn(const int* __restrict__ eidx,
                      const _Float16* __restrict__ qn, const _Float16* __restrict__ kn,
                      const _Float16* __restrict__ vn,
                      const float* __restrict__ Wo, const float* __restrict__ bo,
                      float* __restrict__ agg, float* __restrict__ cnt, int E) {
    const int e = blockIdx.x;
    const int l = threadIdx.x;            // 0..31 : lane == token row
    const int src = eidx[e];
    const int dst = eidx[E + e];

    __shared__ __align__(16) _Float16 Qraw[32][8];     // raw 16B rows (async landing zone)
    __shared__ __align__(16) _Float16 Kraw[32][8];
    __shared__ __align__(16) _Float16 Vraw[32][8];
    __shared__ __align__(16) float    Qp[2][4][32];    // per-head A-frag layout, pad row 3 = 0
    __shared__ __align__(16) float    Kp[2][4][32];    // per-head B-frag layout, pad row 3 = 0
    __shared__ __align__(64) _Float16 Vt[8][32];       // V transposed [dim][k], rows 6..7 = 0
    __shared__ __align__(16) float    Sld[32][32];     // scores
    __shared__ __align__(16) _Float16 attnH[32][32];   // softmaxed attn (f16)
    __shared__ __align__(16) float    msgL[32][8];     // per-token message (6 valid)

    // ---- stage raw Q/K/V rows into LDS
#ifdef AMPGCN_HAVE_ASYNC
    {
        _Float16* gq = const_cast<_Float16*>(qn) + ((size_t)dst * FTOK + l) * 8;
        _Float16* gk = const_cast<_Float16*>(kn) + ((size_t)src * FTOK + l) * 8;
        _Float16* gv = const_cast<_Float16*>(vn) + ((size_t)src * FTOK + l) * 8;
        __builtin_amdgcn_global_load_async_to_lds_b128(
            (GLOBAL_AS v4i*)gq, (LDS_AS v4i*)&Qraw[l][0], 0, 0);
        __builtin_amdgcn_global_load_async_to_lds_b128(
            (GLOBAL_AS v4i*)gk, (LDS_AS v4i*)&Kraw[l][0], 0, 0);
        __builtin_amdgcn_global_load_async_to_lds_b128(
            (GLOBAL_AS v4i*)gv, (LDS_AS v4i*)&Vraw[l][0], 0, 0);
    }
#if __has_builtin(__builtin_amdgcn_s_wait_asynccnt)
    __builtin_amdgcn_s_wait_asynccnt(0);
#else
    asm volatile("s_wait_asynccnt 0" ::: "memory");
#endif
    __syncthreads();
#else
    ((uint4*)Qraw)[l] = ((const uint4*)qn)[(size_t)dst * FTOK + l];
    ((uint4*)Kraw)[l] = ((const uint4*)kn)[(size_t)src * FTOK + l];
    ((uint4*)Vraw)[l] = ((const uint4*)vn)[(size_t)src * FTOK + l];
    __syncthreads();
#endif

    // ---- scatter into WMMA-friendly layouts (branchless fragment builds later)
    {
        union { uint4 u; _Float16 h[8]; } qr, kr, vr;
        qr.u = *(const uint4*)&Qraw[l][0];
        kr.u = *(const uint4*)&Kraw[l][0];
        vr.u = *(const uint4*)&Vraw[l][0];
#pragma unroll
        for (int h = 0; h < 2; ++h) {
#pragma unroll
            for (int d = 0; d < 3; ++d) {
                Qp[h][d][l] = (float)qr.h[3 * h + d];
                Kp[h][d][l] = (float)kr.h[3 * h + d];
            }
            Qp[h][3][l] = 0.f;
            Kp[h][3][l] = 0.f;
        }
#pragma unroll
        for (int d = 0; d < DDIM; ++d) Vt[d][l] = vr.h[d];
        Vt[6][l] = (_Float16)0.f;
        Vt[7][l] = (_Float16)0.f;
    }
    __syncthreads();

    const float rsqrt3 = 0.5773502691896258f;
    const int  rm   = l & 15;
    const bool lo   = l < 16;
    const int  colA = (l >> 4) << 1;          // 0 for lanes 0-15, 2 for lanes 16-31

#pragma unroll
    for (int h = 0; h < 2; ++h) {
        const int bd = 3 * h;

        // ---- scores S = (Q Kt)/sqrt(3): four 16x16 tiles, K=4 (head-dim 3 + zero pad)
#pragma unroll
        for (int i = 0; i < 2; ++i) {
            v2f a;                                    // A 16x4 f32 fragment (branchless)
            a.x = Qp[h][colA][i * 16 + rm];
            a.y = Qp[h][colA + 1][i * 16 + rm];       // pad row 3 is zero for hi lanes
#pragma unroll
            for (int j = 0; j < 2; ++j) {
                v2f bb;                               // B 4x16 f32 fragment: B[k][n]=K[n][k]
                bb.x = Kp[h][colA][j * 16 + rm];
                bb.y = Kp[h][colA + 1][j * 16 + rm];
                v8f c = {};
                c = __builtin_amdgcn_wmma_f32_16x16x4_f32(
                        false, a, false, bb, (short)0, c, false, false);
                float* sp = &Sld[i * 16 + (lo ? 0 : 8)][j * 16 + rm];   // C/D layout
#pragma unroll
                for (int r = 0; r < 8; ++r) sp[r * 32] = c[r] * rsqrt3;
            }
        }
        __syncthreads();

        // ---- softmax: lane l owns query row l (contiguous 128B row, vector loads)
        {
            float rowv[32];
#pragma unroll
            for (int w = 0; w < 8; ++w) {
                float4 f4 = ((const float4*)&Sld[l][0])[w];
                rowv[4 * w + 0] = f4.x; rowv[4 * w + 1] = f4.y;
                rowv[4 * w + 2] = f4.z; rowv[4 * w + 3] = f4.w;
            }
            float mx = rowv[0];
#pragma unroll
            for (int n2 = 1; n2 < 32; ++n2) mx = fmaxf(mx, rowv[n2]);
            float sum = 0.f;
#pragma unroll
            for (int n2 = 0; n2 < 32; ++n2) { rowv[n2] = __expf(rowv[n2] - mx); sum += rowv[n2]; }
            float inv = 1.f / sum;
            uint32_t pk[16];
#pragma unroll
            for (int w = 0; w < 16; ++w) {
                _Float16 h0 = (_Float16)(rowv[2 * w] * inv);
                _Float16 h1 = (_Float16)(rowv[2 * w + 1] * inv);
                pk[w] = (uint32_t)__builtin_bit_cast(unsigned short, h0)
                      | ((uint32_t)__builtin_bit_cast(unsigned short, h1) << 16);
            }
            uint4* ap = (uint4*)&attnH[l][0];
#pragma unroll
            for (int w = 0; w < 4; ++w)
                ap[w] = uint4{pk[4 * w], pk[4 * w + 1], pk[4 * w + 2], pk[4 * w + 3]};
        }
        __syncthreads();

        // ---- msg = attn @ V: two 16x16 tiles, K=32 exactly (f16 WMMA)
        {
            const int byteoff = lo ? 0 : 16;      // A-frag K ranges {0-7,16-23} | {8-15,24-31}
            const int kk      = lo ? 0 : 8;       // B-frag K ranges {0-7,16-23} | {8-15,24-31}
            const int vrow    = (rm < 3) ? (bd + rm) : 6;   // rows 6..7 of Vt are zero
#pragma unroll
            for (int i = 0; i < 2; ++i) {
                union { uint4 u[2]; v16h v; } afr;
                const char* rp = (const char*)&attnH[i * 16 + rm][0];
                afr.u[0] = *(const uint4*)(rp + byteoff);
                afr.u[1] = *(const uint4*)(rp + 32 + byteoff);
                union { uint4 u[2]; v16h v; } bfr;
                const char* vp = (const char*)&Vt[vrow][kk];
                bfr.u[0] = *(const uint4*)(vp);
                bfr.u[1] = *(const uint4*)(vp + 32);
                v8f d = {};
                d = __builtin_amdgcn_wmma_f32_16x16x32_f16(
                        false, afr.v, false, bfr.v, (short)0, d, false, false);
                if (rm < 3) {
                    float* mp = &msgL[i * 16 + (lo ? 0 : 8)][bd + rm];
#pragma unroll
                    for (int r = 0; r < 8; ++r) mp[r * 8] = d[r];
                }
            }
        }
        __syncthreads();
    }

    // ---- output projection [32,6] @ Wo + bo, atomic aggregate at dst
    float m[DDIM];
#pragma unroll
    for (int j = 0; j < DDIM; ++j) m[j] = msgL[l][j];
#pragma unroll
    for (int j = 0; j < DDIM; ++j) {
        float o = bo[j];
#pragma unroll
        for (int i2 = 0; i2 < DDIM; ++i2) o += m[i2] * Wo[i2 * DDIM + j];
        atomicAdd(&agg[(size_t)dst * HID + l * DDIM + j], o);
    }
    if (l == 0) atomicAdd(&cnt[dst], 1.0f);
}

// ---------------------------------------------------------------- mean-divide + BN stats (1 block / channel)
__global__ void ampgcn_bn_stats(float* __restrict__ agg, const float* __restrict__ cnt,
                                float* __restrict__ mu, float* __restrict__ rstd, int Nn) {
    const int c = blockIdx.x;             // 0..191
    const int tid = threadIdx.x;
    float s = 0.f, sq = 0.f;
    for (int n = tid; n < Nn; n += blockDim.x) {
        float ic = 1.f / fmaxf(cnt[n], 1.f);
        float v = agg[(size_t)n * HID + c] * ic;
        agg[(size_t)n * HID + c] = v;
        s += v; sq += v * v;
    }
    __shared__ float s1[256], s2[256];
    s1[tid] = s; s2[tid] = sq;
    __syncthreads();
    for (int off = 128; off > 0; off >>= 1) {
        if (tid < off) { s1[tid] += s1[tid + off]; s2[tid] += s2[tid + off]; }
        __syncthreads();
    }
    if (tid == 0) {
        float m = s1[0] / (float)Nn;
        float v = s2[0] / (float)Nn - m * m;
        mu[c] = m;
        rstd[c] = rsqrtf(v + 1e-5f);
    }
}

// ---------------------------------------------------------------- BN apply + ReLU + Linear + log_softmax
__global__ __launch_bounds__(128)
void ampgcn_classify(const float* __restrict__ hbuf,
                     const float* __restrict__ mu, const float* __restrict__ rstd,
                     const float* __restrict__ gamma, const float* __restrict__ beta,
                     const float* __restrict__ lin_w, const float* __restrict__ lin_b,
                     float* __restrict__ out, int Nn) {
    __shared__ float Wl[NCLS * HID];
    __shared__ float MU[HID], RS[HID], GA[HID], BE[HID];
    for (int i = threadIdx.x; i < NCLS * HID; i += blockDim.x) Wl[i] = lin_w[i];
    for (int i = threadIdx.x; i < HID; i += blockDim.x) {
        MU[i] = mu[i]; RS[i] = rstd[i]; GA[i] = gamma[i]; BE[i] = beta[i];
    }
    __syncthreads();
    int n = blockIdx.x * blockDim.x + threadIdx.x;
    if (n >= Nn) return;

    float lg[NCLS];
#pragma unroll
    for (int j = 0; j < NCLS; ++j) lg[j] = lin_b[j];
    for (int c = 0; c < HID; ++c) {
        float hv = (hbuf[(size_t)n * HID + c] - MU[c]) * RS[c] * GA[c] + BE[c];
        hv = fmaxf(hv, 0.f);
#pragma unroll
        for (int j = 0; j < NCLS; ++j) lg[j] += hv * Wl[j * HID + c];
    }
    float mx = lg[0];
#pragma unroll
    for (int j = 1; j < NCLS; ++j) mx = fmaxf(mx, lg[j]);
    float se = 0.f;
#pragma unroll
    for (int j = 0; j < NCLS; ++j) se += __expf(lg[j] - mx);
    float ls = __logf(se);
#pragma unroll
    for (int j = 0; j < NCLS; ++j) out[(size_t)n * NCLS + j] = lg[j] - mx - ls;
}

// ---------------------------------------------------------------- launcher
extern "C" void kernel_launch(void* const* d_in, const int* in_sizes, int n_in,
                              void* d_out, int out_size, void* d_ws, size_t ws_size,
                              hipStream_t stream) {
    const float* x        = (const float*)d_in[0];
    const int*   eidx     = (const int*)d_in[1];
    const float* feat_emb = (const float*)d_in[2];
    const float* val_w    = (const float*)d_in[3];
    const float* val_b    = (const float*)d_in[4];
    const float* Wq       = (const float*)d_in[5];
    const float* Wk       = (const float*)d_in[6];
    const float* Wv       = (const float*)d_in[7];
    const float* bq       = (const float*)d_in[8];
    const float* bk       = (const float*)d_in[9];
    const float* bv       = (const float*)d_in[10];
    const float* Wo       = (const float*)d_in[11];
    const float* bo       = (const float*)d_in[12];
    const float* gamma    = (const float*)d_in[13];
    const float* beta     = (const float*)d_in[14];
    const float* lin_w    = (const float*)d_in[15];
    const float* lin_b    = (const float*)d_in[16];

    const int Nn = in_sizes[0] / FTOK;     // 50000
    const int E  = in_sizes[1] / 2;        // 100000

    // workspace carve-out (256B aligned slabs)
    char*  ws  = (char*)d_ws;
    size_t off = 0;
    auto carve = [&](size_t bytes) {
        void* p = ws + off;
        off = (off + bytes + 255) & ~(size_t)255;
        return p;
    };
    _Float16* qn  = (_Float16*)carve((size_t)Nn * FTOK * 8 * sizeof(_Float16));
    _Float16* kn  = (_Float16*)carve((size_t)Nn * FTOK * 8 * sizeof(_Float16));
    _Float16* vn  = (_Float16*)carve((size_t)Nn * FTOK * 8 * sizeof(_Float16));
    float*    agg = (float*)carve((size_t)Nn * HID * sizeof(float));
    float*    cnt = (float*)carve((size_t)Nn * sizeof(float));
    float*    mu  = (float*)carve(HID * sizeof(float));
    float*    rsd = (float*)carve(HID * sizeof(float));

    // zero accumulators
    {
        int n1 = Nn * HID;
        ampgcn_zero<<<(n1 + 255) / 256, 256, 0, stream>>>(agg, n1);
        ampgcn_zero<<<(Nn + 255) / 256, 256, 0, stream>>>(cnt, Nn);
    }
    // per-node Q/K/V
    {
        int nt = Nn * FTOK;
        ampgcn_node_proj<<<(nt + 255) / 256, 256, 0, stream>>>(
            x, feat_emb, val_w, val_b, Wq, Wk, Wv, bq, bk, bv, qn, kn, vn, Nn);
    }
    // per-edge attention (1 wave per edge)
    ampgcn_edge_attn<<<E, 32, 0, stream>>>(eidx, qn, kn, vn, Wo, bo, agg, cnt, E);
    // BN stats + mean divide
    ampgcn_bn_stats<<<HID, 256, 0, stream>>>(agg, cnt, mu, rsd, Nn);
    // BN apply + ReLU + classifier + log_softmax
    ampgcn_classify<<<(Nn + 127) / 128, 128, 0, stream>>>(
        agg, mu, rsd, gamma, beta, lin_w, lin_b, (float*)d_out, Nn);
}
